// CrossAttention_7215545057495
// MI455X (gfx1250) — matrix operand order repaired and checked by reading, not compiled, not run on hardware
//
#include <hip/hip_runtime.h>
#include <hip/hip_bf16.h>
#include <math.h>

// ---------------------------------------------------------------------------
// CDNA5 (gfx1250) wave32 WMMA implementation of the cross-attention block.
//   All GEMMs use v_wmma_f32_16x16x32_f16 (f16 in, f32 accumulate).
//   Big GEMMs stage tiles via GLOBAL_LOAD_ASYNC_TO_LDS_B128 (ASYNCcnt),
//   double-buffered in LDS, per CDNA5 ISA.
// Fragment layouts (CDNA5 ISA, 7.12.2):
//   A (16x32 f16): lane L -> row M=L%16 ; elem e -> K = 16*(e/8)+8*(L/16)+e%8
//   B (32x16 f16): lane L -> col N=L%16 ; elem e -> K = 16*(e/8)+8*(L/16)+e%8
//   C (16x16 f32): VGPR r -> row M = r + 8*(L/16) ; col N = L%16
// ---------------------------------------------------------------------------

typedef __attribute__((ext_vector_type(16))) _Float16 v16h;
typedef __attribute__((ext_vector_type(8)))  _Float16 v8h;
typedef __attribute__((ext_vector_type(4)))  _Float16 v4h;
typedef __attribute__((ext_vector_type(8)))  float    v8f;
typedef __attribute__((ext_vector_type(4)))  int      v4i;

#define DEV static __device__ __forceinline__

#if __has_builtin(__builtin_amdgcn_global_load_async_to_lds_b128)
#define ATHENA_ASYNC 1
#else
#define ATHENA_ASYNC 0
#endif

// 16-byte global -> LDS copy (async when available).
DEV void copy16_g2l(_Float16* lds_dst, const _Float16* gsrc) {
#if ATHENA_ASYNC
  __builtin_amdgcn_global_load_async_to_lds_b128(
      (v4i*)const_cast<_Float16*>(gsrc), (v4i*)lds_dst,
      /*offset=*/0, /*cpol=*/0);
#else
  *(v8h*)lds_dst = *(const v8h*)gsrc;
#endif
}

DEV void wait_async_le4() {
#if ATHENA_ASYNC
#if __has_builtin(__builtin_amdgcn_s_wait_asynccnt)
  __builtin_amdgcn_s_wait_asynccnt(4);
#else
  asm volatile("s_wait_asynccnt 0x4" ::: "memory");
#endif
#endif
}

DEV void wait_async_le0() {
#if ATHENA_ASYNC
#if __has_builtin(__builtin_amdgcn_s_wait_asynccnt)
  __builtin_amdgcn_s_wait_asynccnt(0);
#else
  asm volatile("s_wait_asynccnt 0x0" ::: "memory");
#endif
#endif
}

DEV v8f wmma32(v16h a, v16h b, v8f c) {
  return __builtin_amdgcn_wmma_f32_16x16x32_f16(
      /*neg_a=*/false, a, /*neg_b=*/false, b,
      /*c_mod=*/(short)0, c, /*reuse_a=*/false, /*reuse_b=*/false);
}

// A fragment from row-major [M,K] f16 at (row0,k0) baked into `base`, leading dim ld.
DEV v16h load_a_frag(const _Float16* base, int ld) {
  const int lane = threadIdx.x & 31;
  const int r = lane & 15;
  const int h8 = (lane >> 4) * 8;
  const _Float16* p = base + (size_t)r * ld + h8;
  v8h lo = *(const v8h*)(p);        // K = 8*half + 0..7
  v8h hi = *(const v8h*)(p + 16);   // K = 16 + 8*half + 0..7
  v16h o;
#pragma unroll
  for (int i = 0; i < 8; ++i) { o[i] = lo[i]; o[i + 8] = hi[i]; }
  return o;
}

// B fragment: base points at (n0,k0) of a row-major [N,K] matrix (B[k][n] = M[n][k]).
DEV v16h load_b_frag(const _Float16* base, int ld) {
  const int lane = threadIdx.x & 31;
  const int n = lane & 15;
  const int h16 = (lane >> 4) * 8;
  const _Float16* p = base + (size_t)n * ld + h16;
  v8h lo = *(const v8h*)(p);        // K = 8*half + 0..7
  v8h hi = *(const v8h*)(p + 16);   // K = 16 + 8*half + 0..7
  v16h o;
#pragma unroll
  for (int i = 0; i < 8; ++i) { o[i] = lo[i]; o[i + 8] = hi[i]; }
  return o;
}

// ---------------------------------------------------------------------------
// LayerNorm over rows of 1024 f32 -> f16 (and optionally f32) output.
// ---------------------------------------------------------------------------
__global__ __launch_bounds__(256)
void ln_kernel(const float* __restrict__ X, const float* __restrict__ gw,
               const float* __restrict__ bw, _Float16* __restrict__ Yh,
               float* __restrict__ Yf) {
  __shared__ float red1[8];
  __shared__ float red2[8];
  const int row = blockIdx.x;
  const int t = threadIdx.x;
  const float* x = X + (size_t)row * 1024;
  float v[4];
  float s = 0.f;
#pragma unroll
  for (int i = 0; i < 4; ++i) { v[i] = x[t + i * 256]; s += v[i]; }
#pragma unroll
  for (int off = 16; off > 0; off >>= 1) s += __shfl_xor(s, off);
  if ((t & 31) == 0) red1[t >> 5] = s;
  __syncthreads();
  float tot = 0.f;
#pragma unroll
  for (int i = 0; i < 8; ++i) tot += red1[i];
  const float mean = tot * (1.f / 1024.f);
  float s2 = 0.f;
#pragma unroll
  for (int i = 0; i < 4; ++i) { float d = v[i] - mean; s2 += d * d; }
#pragma unroll
  for (int off = 16; off > 0; off >>= 1) s2 += __shfl_xor(s2, off);
  if ((t & 31) == 0) red2[t >> 5] = s2;
  __syncthreads();
  float tot2 = 0.f;
#pragma unroll
  for (int i = 0; i < 8; ++i) tot2 += red2[i];
  const float rstd = rsqrtf(tot2 * (1.f / 1024.f) + 1e-5f);
#pragma unroll
  for (int i = 0; i < 4; ++i) {
    const int c = t + i * 256;
    const float y = (v[i] - mean) * rstd * gw[c] + bw[c];
    Yh[(size_t)row * 1024 + c] = (_Float16)y;
    if (Yf) Yf[(size_t)row * 1024 + c] = y;
  }
}

// ---------------------------------------------------------------------------
// f32 -> f16 convert (n4 = count/4)
// ---------------------------------------------------------------------------
__global__ __launch_bounds__(256)
void cvt_kernel(const float* __restrict__ x, _Float16* __restrict__ y, int n4) {
  const int i = blockIdx.x * 256 + threadIdx.x;
  if (i < n4) {
    const float4 f = ((const float4*)x)[i];
    v4h o = {(_Float16)f.x, (_Float16)f.y, (_Float16)f.z, (_Float16)f.w};
    ((v4h*)y)[i] = o;
  }
}

// ---------------------------------------------------------------------------
// Staged GEMM: Y = X[M,K] * W[N,K]^T, f16 operands, f32 accumulate.
// Block = 256 threads = 8 waves (4 M-waves x 2 N-waves); block tile 128x128,
// wave tile 32x64, K step 32. A/B tiles staged in LDS via async copies,
// double-buffered; padded row stride (40 halves) for bank-conflict-free
// fragment reads.
// MODE 0: f16 row-major store.
// MODE 1: f16 per-batch transposed store (v^T), rows-per-batch = 1024.
// MODE 2: f32 store: out = R + gelu_exact(acc).
// ---------------------------------------------------------------------------
template <int MODE>
__global__ __launch_bounds__(256)
void gemm_staged(const _Float16* __restrict__ X, const _Float16* __restrict__ W,
                 _Float16* __restrict__ Yh, const float* __restrict__ R,
                 float* __restrict__ Yf, int M, int N, int K) {
  constexpr int LDT = 40;  // LDS tile row stride in halves (32 data + 8 pad)
  __shared__ __align__(16) _Float16 At[2][128 * LDT];
  __shared__ __align__(16) _Float16 Bt[2][128 * LDT];

  const int t = threadIdx.x;
  const int lane = t & 31;
  const int wave = t >> 5;
  const int r16 = lane & 15, hl = lane >> 4;
  const int wm = wave >> 1, wn = wave & 1;
  const int mblk = blockIdx.y * 128;
  const int nblk = blockIdx.x * 128;

  // Stage one 128x32 A tile + one 128x32 B tile into LDS buffer `buf`.
  // 512 16-byte chunks per tile; each thread copies 2 chunks of each
  // => 4 async instructions per wave per stage.
  auto stage = [&](int buf, int kk) {
#pragma unroll
    for (int c = 0; c < 2; ++c) {
      const int chunk = t + c * 256;   // 0..511
      const int row = chunk >> 2;      // 0..127
      const int seg = (chunk & 3) * 8; // halves within the 32-wide k slab
      copy16_g2l(&At[buf][row * LDT + seg], X + (size_t)(mblk + row) * K + kk + seg);
      copy16_g2l(&Bt[buf][row * LDT + seg], W + (size_t)(nblk + row) * K + kk + seg);
    }
  };

  v8f acc[2][4];
#pragma unroll
  for (int i = 0; i < 2; ++i)
#pragma unroll
    for (int j = 0; j < 4; ++j) acc[i][j] = (v8f){};

  const int nk = K / 32;
  stage(0, 0);
  for (int k = 0; k < nk; ++k) {
    const int buf = k & 1;
    if (k + 1 < nk) {
      stage(buf ^ 1, (k + 1) * 32);
      wait_async_le4();   // stage k's 4 async ops retired (in-order)
    } else {
      wait_async_le0();
    }
    __syncthreads();
    const v16h a0 = load_a_frag(&At[buf][(wm * 32) * LDT], LDT);
    const v16h a1 = load_a_frag(&At[buf][(wm * 32 + 16) * LDT], LDT);
#pragma unroll
    for (int j = 0; j < 4; ++j) {
      const v16h bf = load_b_frag(&Bt[buf][(wn * 64 + j * 16) * LDT], LDT);
      acc[0][j] = wmma32(a0, bf, acc[0][j]);
      acc[1][j] = wmma32(a1, bf, acc[1][j]);
    }
    __syncthreads();  // all waves done reading `buf` before it is restaged
  }

#pragma unroll
  for (int i = 0; i < 2; ++i)
#pragma unroll
    for (int j = 0; j < 4; ++j)
#pragma unroll
      for (int r = 0; r < 8; ++r) {
        const int m = mblk + wm * 32 + i * 16 + r + 8 * hl;
        const int n = nblk + wn * 64 + j * 16 + r16;
        const float x = acc[i][j][r];
        if (MODE == 0) {
          Yh[(size_t)m * N + n] = (_Float16)x;
        } else if (MODE == 1) {
          const int bb = m >> 10, nr = m & 1023;
          Yh[((size_t)bb << 20) + ((size_t)n << 10) + nr] = (_Float16)x;
        } else {
          const float ge = 0.5f * x * (1.0f + erff(x * 0.70710678118654752f));
          Yf[(size_t)m * N + n] = R[(size_t)m * N + n] + ge;
        }
      }
}

// ---------------------------------------------------------------------------
// Flash attention: per (b,h) head, 64 q-rows per block (4 waves x 16 rows).
//  scores = q.k^T / 32, clip +-1e4, online softmax, O += P.v
//  q,k: row-major f16 [B*1024, 1024];  vt: per-batch transposed f16 [B][1024][1024]
// ---------------------------------------------------------------------------
__global__ __launch_bounds__(128)
void attn_kernel(const _Float16* __restrict__ qh, const _Float16* __restrict__ kh,
                 const _Float16* __restrict__ vth, float* __restrict__ O) {
  const int bh = blockIdx.x;             // 0..63
  const int b = bh >> 4, h = bh & 15;
  const int qt = blockIdx.y;             // 0..15
  const int wave = threadIdx.x >> 5;
  const int lane = threadIdx.x & 31;
  const int r16 = lane & 15, hl = lane >> 4;

  __shared__ __align__(16) _Float16 plds[4][16][64];   // per-wave P tile

  const int qrow = b * 1024 + qt * 64 + wave * 16;
  const v16h qa0 = load_a_frag(qh + (size_t)qrow * 1024 + h * 64, 1024);
  const v16h qa1 = load_a_frag(qh + (size_t)qrow * 1024 + h * 64 + 32, 1024);

  float mi[8], li[8];
  v8f o[4];
#pragma unroll
  for (int r = 0; r < 8; ++r) { mi[r] = -1e30f; li[r] = 0.f; }
#pragma unroll
  for (int d = 0; d < 4; ++d) o[d] = (v8f){};

  for (int kt = 0; kt < 16; ++kt) {
    v8f s[4];
#pragma unroll
    for (int j = 0; j < 4; ++j) {
      const _Float16* kb =
          kh + (size_t)(b * 1024 + kt * 64 + j * 16) * 1024 + h * 64;
      v8f z = (v8f){};
      z = wmma32(qa0, load_b_frag(kb, 1024), z);
      z = wmma32(qa1, load_b_frag(kb + 32, 1024), z);
      s[j] = z;
    }
    // scale by 1/sqrt(1024) and clip to +-1e4
#pragma unroll
    for (int j = 0; j < 4; ++j)
#pragma unroll
      for (int r = 0; r < 8; ++r) {
        float x = s[j][r] * 0.03125f;
        s[j][r] = fminf(fmaxf(x, -10000.f), 10000.f);
      }
    // online softmax per row (row = r + 8*hl lives in a 16-lane half-wave)
    float corr[8];
#pragma unroll
    for (int r = 0; r < 8; ++r) {
      float rm = fmaxf(fmaxf(s[0][r], s[1][r]), fmaxf(s[2][r], s[3][r]));
      rm = fmaxf(rm, __shfl_xor(rm, 1));
      rm = fmaxf(rm, __shfl_xor(rm, 2));
      rm = fmaxf(rm, __shfl_xor(rm, 4));
      rm = fmaxf(rm, __shfl_xor(rm, 8));
      const float mn = fmaxf(mi[r], rm);
      corr[r] = __expf(mi[r] - mn);
      mi[r] = mn;
      float rs = 0.f;
#pragma unroll
      for (int j = 0; j < 4; ++j) {
        const float p = __expf(s[j][r] - mn);
        s[j][r] = p;
        rs += p;
      }
      rs += __shfl_xor(rs, 1);
      rs += __shfl_xor(rs, 2);
      rs += __shfl_xor(rs, 4);
      rs += __shfl_xor(rs, 8);
      li[r] = li[r] * corr[r] + rs;
    }
#pragma unroll
    for (int d = 0; d < 4; ++d)
#pragma unroll
      for (int r = 0; r < 8; ++r) o[d][r] *= corr[r];

    // spill P (C layout) to LDS, reload as A fragments
#pragma unroll
    for (int j = 0; j < 4; ++j)
#pragma unroll
      for (int r = 0; r < 8; ++r)
        plds[wave][r + 8 * hl][j * 16 + r16] = (_Float16)s[j][r];
    __syncthreads();
    const v16h pa0 = load_a_frag(&plds[wave][0][0], 64);
    const v16h pa1 = load_a_frag(&plds[wave][0][0] + 32, 64);
#pragma unroll
    for (int d = 0; d < 4; ++d) {
      const _Float16* vb =
          vth + ((size_t)(b * 1024) + h * 64 + d * 16) * 1024 + kt * 64;
      o[d] = wmma32(pa0, load_b_frag(vb, 1024), o[d]);
      o[d] = wmma32(pa1, load_b_frag(vb + 32, 1024), o[d]);
    }
    __syncthreads();
  }

#pragma unroll
  for (int d = 0; d < 4; ++d)
#pragma unroll
    for (int r = 0; r < 8; ++r) {
      const float inv = 1.f / li[r];
      O[(size_t)(qrow + r + 8 * hl) * 1024 + h * 64 + d * 16 + r16] =
          o[d][r] * inv;
    }
}

// ---------------------------------------------------------------------------
// Host-side orchestration
// ---------------------------------------------------------------------------
extern "C" void kernel_launch(void* const* d_in, const int* in_sizes, int n_in,
                              void* d_out, int out_size, void* d_ws, size_t ws_size,
                              hipStream_t stream) {
  (void)in_sizes; (void)n_in; (void)out_size; (void)ws_size;
  const float* Q     = (const float*)d_in[0];
  const float* K     = (const float*)d_in[1];
  const float* V     = (const float*)d_in[2];
  const float* Wq    = (const float*)d_in[3];
  const float* Wk    = (const float*)d_in[4];
  const float* Wv    = (const float*)d_in[5];
  const float* Wo    = (const float*)d_in[6];
  const float* pre_g = (const float*)d_in[7];
  const float* pre_b = (const float*)d_in[8];
  const float* ln_g  = (const float*)d_in[9];
  const float* ln_b  = (const float*)d_in[10];
  float* out = (float*)d_out;

  char* ws = (char*)d_ws;
  size_t off = 0;
  auto alloc = [&](size_t bytes) -> void* {
    void* p = ws + off;
    off += (bytes + 255) & ~(size_t)255;
    return p;
  };
  const size_t MB = 4096, D = 1024;
  _Float16* Qn  = (_Float16*)alloc(MB * D * 2);
  _Float16* Kn  = (_Float16*)alloc(MB * D * 2);
  _Float16* Vh  = (_Float16*)alloc(MB * D * 2);
  _Float16* Wqh = (_Float16*)alloc(D * D * 2);
  _Float16* Wkh = (_Float16*)alloc(D * D * 2);
  _Float16* Wvh = (_Float16*)alloc(D * D * 2);
  _Float16* Woh = (_Float16*)alloc(D * D * 2);
  _Float16* qh  = (_Float16*)alloc(MB * D * 2);
  _Float16* khb = (_Float16*)alloc(MB * D * 2);
  _Float16* vth = (_Float16*)alloc(MB * D * 2);
  float*    Of  = (float*)alloc(MB * D * 4);
  float*    Onf = (float*)alloc(MB * D * 4);
  _Float16* Onh = (_Float16*)alloc(MB * D * 2);

  // 1) LayerNorm Q, K (shared params) -> f16
  ln_kernel<<<4096, 256, 0, stream>>>(Q, pre_g, pre_b, Qn, nullptr);
  ln_kernel<<<4096, 256, 0, stream>>>(K, pre_g, pre_b, Kn, nullptr);

  // 2) f32 -> f16 converts
  cvt_kernel<<<(int)(MB * D / 4 / 256), 256, 0, stream>>>(V, Vh, (int)(MB * D / 4));
  cvt_kernel<<<(int)(D * D / 4 / 256), 256, 0, stream>>>(Wq, Wqh, (int)(D * D / 4));
  cvt_kernel<<<(int)(D * D / 4 / 256), 256, 0, stream>>>(Wk, Wkh, (int)(D * D / 4));
  cvt_kernel<<<(int)(D * D / 4 / 256), 256, 0, stream>>>(Wv, Wvh, (int)(D * D / 4));
  cvt_kernel<<<(int)(D * D / 4 / 256), 256, 0, stream>>>(Wo, Woh, (int)(D * D / 4));

  // 3) Projections (async-staged WMMA GEMMs). v stored per-batch transposed.
  gemm_staged<0><<<dim3(8, 32), 256, 0, stream>>>(Qn, Wqh, qh, nullptr, nullptr,
                                                  4096, 1024, 1024);
  gemm_staged<0><<<dim3(8, 32), 256, 0, stream>>>(Kn, Wkh, khb, nullptr, nullptr,
                                                  4096, 1024, 1024);
  gemm_staged<1><<<dim3(8, 32), 256, 0, stream>>>(Vh, Wvh, vth, nullptr, nullptr,
                                                  4096, 1024, 1024);

  // 4) Flash attention (WMMA), O in f32
  attn_kernel<<<dim3(64, 16), 128, 0, stream>>>(qh, khb, vth, Of);

  // 5) LayerNorm O -> f16 + f32
  ln_kernel<<<4096, 256, 0, stream>>>(Of, ln_g, ln_b, Onh, Onf);

  // 6) out = On + gelu(On @ Wo^T)  (WMMA + erf epilogue)
  gemm_staged<2><<<dim3(8, 32), 256, 0, stream>>>(Onh, Woh, nullptr, Onf, out,
                                                  4096, 1024, 1024);
}
